// CrossAttention_68152541053564
// MI455X (gfx1250) — matrix operand order repaired and checked by reading, not compiled, not run on hardware
//
#include <hip/hip_runtime.h>

// ---------------------------------------------------------------------------
// CrossAttention on MI455X (gfx1250, wave32, WMMA).
//   B=16, S=2048, QD=512, KD=VD=HD=256
// Pipeline:
//   0) pack_wT:  W[Kd,256] f32 -> Wt[256,Kd] f16 (one-time repack; makes all
//                B-fragments contiguous 16B loads)
//   1) proj_gemm: Q/K/V projections f32 -> f16 workspace (WMMA f16)
//   2) flash_attn: streaming softmax(Q K^T) V; K tile moved to LDS via
//                  global_load_async_to_lds_b128 (ASYNCcnt), V tile
//                  transposed into LDS; f16 WMMA, f32 accum
//   3) out_gemm: attended @ Wo + bo -> f32 output (WMMA f16)
// Roofline: ~90 GFLOP vs ~160MB DRAM -> compute-bound on f16 WMMA; every
// matmul must be v_wmma_f32_16x16x32_f16 and the 268MB score tensor must
// never touch memory (flash streaming).
// ---------------------------------------------------------------------------

#define BATCH 16
#define SEQ   2048
#define QDIM  512
#define HDIM  256
#define SCALE 0.0625f   // HD^-0.5 = 1/16

typedef __attribute__((ext_vector_type(16))) _Float16 v16h;
typedef __attribute__((ext_vector_type(8)))  _Float16 v8h;
typedef __attribute__((ext_vector_type(8)))  float    v8f;

#define WMMA_F16(a, b, c) \
  __builtin_amdgcn_wmma_f32_16x16x32_f16(false, (a), false, (b), (short)0, (c), false, false)

// ----- fragment loaders ----------------------------------------------------
// A fragment (16x32 f16): lane m = lane%16 owns row m.
// element e: K = g*16 + e, g = lane/16.
__device__ __forceinline__ v16h ld_a_f16(const _Float16* p /*row base + k0*/, int g) {
  v8h lo = *(const v8h*)(p + g * 16);
  v8h hi = *(const v8h*)(p + g * 16 + 8);
  v16h a;
#pragma unroll
  for (int e = 0; e < 8; ++e) { a[e] = lo[e]; a[8 + e] = hi[e]; }
  return a;
}

__device__ __forceinline__ v16h ld_a_f32(const float* p /*row base + k0*/, int g) {
  v16h a;
#pragma unroll
  for (int e = 0; e < 8; ++e) a[e] = (_Float16)p[g * 16 + e];
#pragma unroll
  for (int e = 0; e < 8; ++e) a[8 + e] = (_Float16)p[g * 16 + 8 + e];
  return a;
}

// B fragment (32x16): lane n = lane%16 owns column n; element e: K = g*16+e.
// p points at the 16 contiguous K-values for this lane.
__device__ __forceinline__ v16h ld_b16(const _Float16* p) {
  v8h lo = *(const v8h*)p;
  v8h hi = *(const v8h*)(p + 8);
  v16h b;
#pragma unroll
  for (int e = 0; e < 8; ++e) { b[e] = lo[e]; b[8 + e] = hi[e]; }
  return b;
}

// ----- kernel 0: weight repack  Wt[col][k] = (f16) W[k][col] ---------------
__global__ __launch_bounds__(256) void pack_wT(
    const float* __restrict__ W, _Float16* __restrict__ Wt, int Kd) {
  const int col = blockIdx.x;                 // 0..255 output column
  for (int k = threadIdx.x; k < Kd; k += blockDim.x)
    Wt[(size_t)col * Kd + k] = (_Float16)W[(size_t)k * HDIM + col];
}

// ----- kernel 1: Y[M,256] = X[M,Kd] @ W + bias, f16 out; W pre-packed ------
__global__ __launch_bounds__(128) void proj_gemm(
    const float* __restrict__ X, const _Float16* __restrict__ Wt /*[256][Kd]*/,
    const float* __restrict__ bias, _Float16* __restrict__ Y, int Kd) {
  const int lane = threadIdx.x & 31;
  const int wave = threadIdx.x >> 5;
  const int n = lane & 15;
  const int g = lane >> 4;
  const int row0 = blockIdx.x * 16;
  const int col0 = blockIdx.y * 64 + wave * 16;

  const float* arow = X + (size_t)(row0 + n) * Kd;          // lane m == n
  const _Float16* bcol = Wt + (size_t)(col0 + n) * Kd;      // contiguous K

  v8f acc = {};
  for (int k0 = 0; k0 < Kd; k0 += 32) {
    v16h a = ld_a_f32(arow + k0, g);
    v16h b = ld_b16(bcol + k0 + g * 16);
    acc = WMMA_F16(a, b, acc);
  }
  const float bv = bias[col0 + n];
#pragma unroll
  for (int r = 0; r < 8; ++r) {
    const int row = row0 + r + 8 * g;
    Y[(size_t)row * HDIM + col0 + n] = (_Float16)(acc[r] + bv);
  }
}

// ----- kernel 2: flash attention ------------------------------------------
// Block: 128 threads (4 waves), 64 query rows of one batch.
// 32-key tiles: K tile -> LDS via async-to-LDS (ASYNCcnt), V tile transposed.
__global__ __launch_bounds__(128) void flash_attn(
    const _Float16* __restrict__ Q, const _Float16* __restrict__ K,
    const _Float16* __restrict__ V, _Float16* __restrict__ O) {
  __shared__ _Float16 Kt[32][HDIM];      // 16 KB, Kt[key][h]
  __shared__ _Float16 Vt[HDIM][32];      // 16 KB, Vt[h][key]  (transposed)
  __shared__ _Float16 Pst[4][16][32];    //  4 KB, per-wave P staging

  const int lane = threadIdx.x & 31;
  const int wave = threadIdx.x >> 5;
  const int n = lane & 15;
  const int g = lane >> 4;
  const int b = blockIdx.y;
  const int q0 = blockIdx.x * 64 + wave * 16;

  const _Float16* Qb = Q + (size_t)b * SEQ * HDIM;
  const _Float16* Kb = K + (size_t)b * SEQ * HDIM;
  const _Float16* Vb = V + (size_t)b * SEQ * HDIM;

  // Q fragments for this wave's 16 rows, full HD=256 (8 chunks of K=32).
  v16h qf[8];
  const _Float16* qrow = Qb + (size_t)(q0 + n) * HDIM;
#pragma unroll
  for (int h = 0; h < 8; ++h) qf[h] = ld_a_f16(qrow + h * 32, g);

  v8f zero = {};
  v8f o[16];
#pragma unroll
  for (int i = 0; i < 16; ++i) o[i] = zero;
  float mstat[8], lstat[8];
#pragma unroll
  for (int r = 0; r < 8; ++r) { mstat[r] = -__builtin_inff(); lstat[r] = 0.0f; }

  // cooperative copy coordinates: thread -> (key row, 64-half column chunk)
  const int ckey = threadIdx.x >> 2;
  const int ccol = (threadIdx.x & 3) * 64;
  // LDS byte offset of this thread's K-tile destination (low 32 bits of the
  // generic pointer to LDS == DS offset on AMDGPU).
  const unsigned kt_lds = (unsigned)(unsigned long long)(uintptr_t)&Kt[ckey][ccol];

  for (int kt = 0; kt < SEQ; kt += 32) {
    __syncthreads();  // previous iteration's LDS reads complete
    {
      const _Float16* ksrc = Kb + (size_t)(kt + ckey) * HDIM + ccol;
      const _Float16* vsrc = Vb + (size_t)(kt + ckey) * HDIM + ccol;
      // K tile: 8 x 16B async copies straight into LDS (no VGPR round-trip).
      const unsigned long long kga = (unsigned long long)(uintptr_t)ksrc;
#pragma unroll
      for (int i = 0; i < 8; ++i) {
        asm volatile("global_load_async_to_lds_b128 %0, %1, off"
                     :: "v"(kt_lds + i * 16), "v"(kga + i * 16) : "memory");
      }
      // V tile: vector loads + transposed scatter into LDS.
#pragma unroll
      for (int i = 0; i < 8; ++i) {
        v8h vv = *(const v8h*)(vsrc + i * 8);
#pragma unroll
        for (int j = 0; j < 8; ++j) Vt[ccol + i * 8 + j][ckey] = vv[j];
      }
      // prefetch next tile while this one is consumed
      if (kt + 32 < SEQ) {
        __builtin_prefetch(Kb + (size_t)(kt + 32 + ckey) * HDIM + ccol, 0, 3);
        __builtin_prefetch(Vb + (size_t)(kt + 32 + ckey) * HDIM + ccol, 0, 3);
      }
    }
    asm volatile("s_wait_asynccnt 0x0" ::: "memory");
    __syncthreads();

    // scores: two 16x16 tiles covering 32 keys
    v8f s0 = zero, s1 = zero;
#pragma unroll
    for (int h = 0; h < 8; ++h) {
      v16h b0 = ld_b16(&Kt[n][h * 32 + g * 16]);
      v16h b1 = ld_b16(&Kt[16 + n][h * 32 + g * 16]);
      s0 = WMMA_F16(qf[h], b0, s0);
      s1 = WMMA_F16(qf[h], b1, s1);
    }

    // online softmax (row stats shared by the 16 lanes of each half-wave)
    float p0[8], p1[8], alpha[8];
#pragma unroll
    for (int r = 0; r < 8; ++r) {
      float x0 = s0[r] * SCALE;
      float x1 = s1[r] * SCALE;
      float t = fmaxf(x0, x1);
#pragma unroll
      for (int off = 1; off < 16; off <<= 1) t = fmaxf(t, __shfl_xor(t, off));
      float mn = fmaxf(mstat[r], t);
      float a_ = __expf(mstat[r] - mn);
      float e0 = __expf(x0 - mn);
      float e1 = __expf(x1 - mn);
      float rs = e0 + e1;
#pragma unroll
      for (int off = 1; off < 16; off <<= 1) rs += __shfl_xor(rs, off);
      lstat[r] = lstat[r] * a_ + rs;
      mstat[r] = mn;
      p0[r] = e0; p1[r] = e1; alpha[r] = a_;
    }
#pragma unroll
    for (int ht = 0; ht < 16; ++ht)
#pragma unroll
      for (int r = 0; r < 8; ++r) o[ht][r] *= alpha[r];

    // stage P (C-layout -> row-major) and reload in A-layout
#pragma unroll
    for (int r = 0; r < 8; ++r) {
      Pst[wave][r + 8 * g][n]      = (_Float16)p0[r];
      Pst[wave][r + 8 * g][16 + n] = (_Float16)p1[r];
    }
    __syncthreads();
    v16h pa = ld_a_f16(&Pst[wave][n][0], g);

    // accumulate P @ V over all 16 h-tiles
#pragma unroll
    for (int ht = 0; ht < 16; ++ht) {
      v16h vb = ld_b16(&Vt[ht * 16 + n][g * 16]);
      o[ht] = WMMA_F16(pa, vb, o[ht]);
    }
  }

  // epilogue: divide by softmax denominator, store attended as f16
  float inv[8];
#pragma unroll
  for (int r = 0; r < 8; ++r) inv[r] = 1.0f / lstat[r];
#pragma unroll
  for (int ht = 0; ht < 16; ++ht) {
#pragma unroll
    for (int r = 0; r < 8; ++r) {
      const int row = q0 + r + 8 * g;
      O[((size_t)b * SEQ + row) * HDIM + ht * 16 + n] = (_Float16)(o[ht][r] * inv[r]);
    }
  }
}

// ----- kernel 3: out[M,256] = A[M,256](f16) @ Wo + bo, f32 out -------------
__global__ __launch_bounds__(128) void out_gemm(
    const _Float16* __restrict__ A, const _Float16* __restrict__ Wt /*[256][256]*/,
    const float* __restrict__ bias, float* __restrict__ Y) {
  const int lane = threadIdx.x & 31;
  const int wave = threadIdx.x >> 5;
  const int n = lane & 15;
  const int g = lane >> 4;
  const int row0 = blockIdx.x * 16;
  const int col0 = blockIdx.y * 64 + wave * 16;

  const _Float16* arow = A + (size_t)(row0 + n) * HDIM;
  const _Float16* bcol = Wt + (size_t)(col0 + n) * HDIM;

  v8f acc = {};
#pragma unroll
  for (int k0 = 0; k0 < HDIM; k0 += 32) {
    v16h a = ld_a_f16(arow + k0, g);
    v16h b = ld_b16(bcol + k0 + g * 16);
    acc = WMMA_F16(a, b, acc);
  }
  const float bv = bias[col0 + n];
#pragma unroll
  for (int r = 0; r < 8; ++r) {
    const int row = row0 + r + 8 * g;
    Y[(size_t)row * HDIM + col0 + n] = acc[r] + bv;
  }
}

// ---------------------------------------------------------------------------
extern "C" void kernel_launch(void* const* d_in, const int* in_sizes, int n_in,
                              void* d_out, int out_size, void* d_ws, size_t ws_size,
                              hipStream_t stream) {
  const float* query = (const float*)d_in[0];
  const float* key   = (const float*)d_in[1];
  const float* value = (const float*)d_in[2];
  const float* Wq    = (const float*)d_in[3];
  const float* bq    = (const float*)d_in[4];
  const float* Wk    = (const float*)d_in[5];
  const float* bk    = (const float*)d_in[6];
  const float* Wv    = (const float*)d_in[7];
  const float* bv    = (const float*)d_in[8];
  const float* Wo    = (const float*)d_in[9];
  const float* bo    = (const float*)d_in[10];
  float* out = (float*)d_out;

  const size_t elems = (size_t)BATCH * SEQ * HDIM;   // 8Mi f16 elements each
  _Float16* Qws = (_Float16*)d_ws;
  _Float16* Kws = Qws + elems;
  _Float16* Vws = Kws + elems;
  _Float16* Aws = Vws + elems;
  _Float16* WqT = Aws + elems;                       // [256][512]
  _Float16* WkT = WqT + (size_t)HDIM * QDIM;         // [256][256]
  _Float16* WvT = WkT + (size_t)HDIM * HDIM;
  _Float16* WoT = WvT + (size_t)HDIM * HDIM;

  // one-time weight repacks (f32 row-major -> f16 transposed)
  pack_wT<<<dim3(HDIM), dim3(256), 0, stream>>>(Wq, WqT, QDIM);
  pack_wT<<<dim3(HDIM), dim3(256), 0, stream>>>(Wk, WkT, HDIM);
  pack_wT<<<dim3(HDIM), dim3(256), 0, stream>>>(Wv, WvT, HDIM);
  pack_wT<<<dim3(HDIM), dim3(256), 0, stream>>>(Wo, WoT, HDIM);

  const int M = BATCH * SEQ;                // 32768 rows
  dim3 gemm_grid(M / 16, HDIM / 64);        // (2048, 4)
  dim3 gemm_block(128);

  proj_gemm<<<gemm_grid, gemm_block, 0, stream>>>(query, WqT, bq, Qws, QDIM);
  proj_gemm<<<gemm_grid, gemm_block, 0, stream>>>(key,   WkT, bk, Kws, HDIM);
  proj_gemm<<<gemm_grid, gemm_block, 0, stream>>>(value, WvT, bv, Vws, HDIM);

  dim3 fa_grid(SEQ / 64, BATCH);            // (32, 16)
  flash_attn<<<fa_grid, gemm_block, 0, stream>>>(Qws, Kws, Vws, Aws);

  out_gemm<<<gemm_grid, gemm_block, 0, stream>>>(Aws, WoT, bo, out);
}
